// MultiHeadAttention_8942121911078
// MI455X (gfx1250) — compile-verified, matrix operand order checked by reading, not run on hardware
//
#include <hip/hip_runtime.h>

typedef __attribute__((ext_vector_type(16))) __bf16 v16bf;
typedef __attribute__((ext_vector_type(8)))  __bf16 v8bf;
typedef __attribute__((ext_vector_type(8)))  float  v8f;
typedef __attribute__((ext_vector_type(4)))  int    v4i;

#define B_ 4
#define T_ 2048
#define C_ 1024
#define H_ 16
#define D_ 64
#define BT_ (B_*T_)

// LDS row stride (in bf16 elements) for staged 32-wide K tiles: 80 bytes keeps
// 16B alignment for b128 accesses while avoiding 4-way bank conflicts.
#define LDSTRIDE 40

#if defined(__has_builtin)
# if __has_builtin(__builtin_amdgcn_global_load_async_to_lds_b128)
#  define HAVE_ASYNC_LDS 1
# endif
#endif
#ifndef HAVE_ASYNC_LDS
# define HAVE_ASYNC_LDS 0
#endif

#if HAVE_ASYNC_LDS
typedef __attribute__((address_space(1))) v4i g_v4i;   // global <4 x i32>
typedef __attribute__((address_space(3))) v4i l_v4i;   // LDS    <4 x i32>
#endif

// Build a 16x32 bf16 WMMA fragment (A-layout): lane holds one row, the 16
// elements are K = 16*g + 8*half + j  (g=0..1, j=0..7). Caller passes
// p = rowBase + k0 + half*8 ; elements [0..7] from p, [8..15] from p+16.
__device__ __forceinline__ v16bf load_frag(const __bf16* p) {
    v8bf lo = *(const v8bf*)(p);
    v8bf hi = *(const v8bf*)(p + 16);
    return __builtin_shufflevector(lo, hi, 0,1,2,3,4,5,6,7,8,9,10,11,12,13,14,15);
}

__device__ __forceinline__ v8f wmma_bf16(v16bf a, v16bf b, v8f c) {
    return __builtin_amdgcn_wmma_f32_16x16x32_bf16(false, a, false, b, (short)0, c, false, false);
}

// Stage a 64-row x 32-K bf16 tile into LDS (row stride LDSTRIDE).
// gsrc_base points at (row 0, k0); global row stride is C_ elements.
// 256 16-byte chunks, 128 threads -> 2 chunks each.
__device__ __forceinline__ void stage_tile(__bf16* lds_dst, const __bf16* gsrc_base, int tid) {
#pragma unroll
    for (int c = tid; c < 256; c += 128) {
        int row = c >> 2, seg = c & 3;
        const __bf16* g = gsrc_base + (size_t)row * C_ + seg * 8;
        __bf16* l = lds_dst + row * LDSTRIDE + seg * 8;
#if HAVE_ASYNC_LDS
        __builtin_amdgcn_global_load_async_to_lds_b128((g_v4i*)g, (l_v4i*)l, 0, 0);
#else
        *(v8bf*)l = *(const v8bf*)g;
#endif
    }
}

__device__ __forceinline__ void wait_stage() {
#if HAVE_ASYNC_LDS
# if __has_builtin(__builtin_amdgcn_s_wait_asynccnt)
    __builtin_amdgcn_s_wait_asynccnt(0);
# else
    asm volatile("s_wait_asynccnt 0" ::: "memory");
# endif
#endif
}

// ---------------- conversion kernels ----------------

__global__ void cvt_x_kernel(const float* __restrict__ x, __bf16* __restrict__ xb, int n) {
    int i = blockIdx.x * 256 + threadIdx.x;
    if (i < n) xb[i] = (__bf16)x[i];
}

// w: [H, C, D] (f32)  ->  wT: [H, D, C] (bf16)
__global__ void cvt_whcd_kernel(const float* __restrict__ w, __bf16* __restrict__ wT) {
    int i = blockIdx.x * 256 + threadIdx.x;       // over H*C*D
    int h = i / (C_ * D_);
    int r = i - h * (C_ * D_);
    int c = r / D_;
    int d = r - c * D_;
    wT[((size_t)h * D_ + d) * C_ + c] = (__bf16)w[i];
}

// wp: [C_in, C_out] (f32) -> wpT: [C_out, C_in] (bf16)
__global__ void cvt_wp_kernel(const float* __restrict__ w, __bf16* __restrict__ wT) {
    int i = blockIdx.x * 256 + threadIdx.x;       // over C*C
    int ci = i / C_;
    int co = i - ci * C_;
    wT[(size_t)co * C_ + ci] = (__bf16)w[i];
}

// ---------------- QKV projection ----------------
// grid: (BT/64, H, 3)  block: 128 (4 waves).  B (weight) tile staged in LDS,
// double-buffered with async copies; A fragments register-pipelined.
__global__ __launch_bounds__(128) void qkv_kernel(
    const __bf16* __restrict__ xb,
    const __bf16* __restrict__ wqT, const __bf16* __restrict__ wkT, const __bf16* __restrict__ wvT,
    __bf16* __restrict__ qb, __bf16* __restrict__ kb, __bf16* __restrict__ vtb)
{
    __shared__ __align__(16) __bf16 lds_b[2 * 64 * LDSTRIDE];

    const int tid  = threadIdx.x;
    const int wave = tid >> 5;
    const int lane = tid & 31;
    const int half = lane >> 4;
    const int lr   = lane & 15;
    const int h    = blockIdx.y;
    const int z    = blockIdx.z;
    const int rbase = blockIdx.x * 64 + wave * 16;

    const __bf16* wT = (z == 0) ? wqT : (z == 1) ? wkT : wvT;
    const __bf16* wbase = wT + (size_t)h * D_ * C_;          // 64 rows x C_
    const __bf16* arow  = xb + (size_t)(rbase + lr) * C_ + half * 8;

    v8f acc[4] = {};

    stage_tile(lds_b, wbase, tid);              // k0 = 0 into buffer 0
    v16bf a_cur = load_frag(arow);

    for (int k0 = 0; k0 < C_; k0 += 32) {
        const int cb = (k0 >> 5) & 1;
        wait_stage();                           // our async copies for buf cb done
        __syncthreads();                        // published; prior reads of buf cb^1 done
        if (k0 + 32 < C_)
            stage_tile(lds_b + (cb ^ 1) * (64 * LDSTRIDE), wbase + k0 + 32, tid);
        v16bf a_nxt = a_cur;
        if (k0 + 32 < C_) a_nxt = load_frag(arow + k0 + 32);

        const __bf16* lb = lds_b + cb * (64 * LDSTRIDE);
#pragma unroll
        for (int nt = 0; nt < 4; ++nt) {
            v16bf b = load_frag(lb + (nt * 16 + lr) * LDSTRIDE + half * 8);
            acc[nt] = wmma_bf16(a_cur, b, acc[nt]);
        }
        a_cur = a_nxt;
    }

    // D layout: lane -> col n = lr, element e -> row m = e + 8*half
#pragma unroll
    for (int nt = 0; nt < 4; ++nt) {
#pragma unroll
        for (int e = 0; e < 8; ++e) {
            int row = rbase + e + 8 * half;       // flat B*T row
            int bb  = row >> 11;                  // /T_
            int t   = row & (T_ - 1);
            int d   = nt * 16 + lr;
            float val = acc[nt][e];
            if (z == 2) {
                vtb[(((size_t)bb * H_ + h) * D_ + d) * T_ + t] = (__bf16)val;  // [B,H,D,T]
            } else {
                __bf16* outp = (z == 0) ? qb : kb;                              // [B,H,T,D]
                outp[(((size_t)bb * H_ + h) * T_ + t) * D_ + d] = (__bf16)val;
            }
        }
    }
}

// ---------------- flash attention ----------------
// grid: (T/64, H, B)  block: 128.  Each wave: 16 q-rows, keys streamed 32 at a
// time; V loads issued before softmax, next K fragments prefetched during it.
__global__ __launch_bounds__(128) void flash_kernel(
    const __bf16* __restrict__ qb, const __bf16* __restrict__ kb,
    const __bf16* __restrict__ vtb, __bf16* __restrict__ ob)
{
    __shared__ __align__(16) __bf16 lds_p[4 * 16 * LDSTRIDE];   // per-wave P tile

    const int wave = threadIdx.x >> 5;
    const int lane = threadIdx.x & 31;
    const int half = lane >> 4;
    const int lr   = lane & 15;
    const int h    = blockIdx.y;
    const int b    = blockIdx.z;
    const int wq0  = blockIdx.x * 64 + wave * 16;

    const size_t bh = (size_t)b * H_ + h;
    const __bf16* kbase = kb + bh * T_ * D_;
    const __bf16* vbase = vtb + bh * D_ * T_;

    const __bf16* qrow = qb + (bh * T_ + wq0 + lr) * D_ + half * 8;
    v16bf aq0 = load_frag(qrow);
    v16bf aq1 = load_frag(qrow + 32);

    float m_i[8], l_i[8];
    v8f oacc[4] = {};
#pragma unroll
    for (int e = 0; e < 8; ++e) { m_i[e] = -1e30f; l_i[e] = 0.0f; }

    __bf16* pbuf = lds_p + wave * 16 * LDSTRIDE;
    const int send = wq0 + 16;   // causal: last needed key index + 1

    auto load_kfrags = [&](int s0, v16bf kf[4]) {
#pragma unroll
        for (int j = 0; j < 2; ++j) {
            int krow = s0 + 16 * j + lr;
            int krc  = krow < T_ ? krow : T_ - 1;      // clamp (masked anyway)
            const __bf16* kp = kbase + (size_t)krc * D_ + half * 8;
            kf[2 * j]     = load_frag(kp);
            kf[2 * j + 1] = load_frag(kp + 32);
        }
    };

    v16bf kf[4];
    load_kfrags(0, kf);

    for (int s0 = 0; s0 < send; s0 += 32) {
        // S = Q * K^T for 16 q-rows x 32 keys
        v8f sacc[2] = {};
        sacc[0] = wmma_bf16(aq0, kf[0], sacc[0]);
        sacc[0] = wmma_bf16(aq1, kf[1], sacc[0]);
        sacc[1] = wmma_bf16(aq0, kf[2], sacc[1]);
        sacc[1] = wmma_bf16(aq1, kf[3], sacc[1]);

        // issue V loads now so they overlap the softmax chain
        v16bf vf[4];
#pragma unroll
        for (int nt = 0; nt < 4; ++nt)
            vf[nt] = load_frag(vbase + (size_t)(nt * 16 + lr) * T_ + s0 + half * 8);

        // prefetch next iteration's K fragments during softmax
        v16bf kn[4] = {kf[0], kf[1], kf[2], kf[3]};
        if (s0 + 32 < send) load_kfrags(s0 + 32, kn);

        // scale + causal mask.  D layout: col(key) = lr, row(q) = e + 8*half
        float ps[2][8];
#pragma unroll
        for (int j = 0; j < 2; ++j) {
            int key = s0 + 16 * j + lr;
#pragma unroll
            for (int e = 0; e < 8; ++e) {
                int qi = wq0 + e + 8 * half;
                float v = sacc[j][e] * 0.125f;      // 1/sqrt(64)
                ps[j][e] = (key <= qi) ? v : -1e30f;
            }
        }
        // online softmax: one score row lives on 16 lanes of one half-wave
#pragma unroll
        for (int e = 0; e < 8; ++e) {
            float mx = fmaxf(ps[0][e], ps[1][e]);
            for (int off = 1; off < 16; off <<= 1)
                mx = fmaxf(mx, __shfl_xor(mx, off));
            float nm    = fmaxf(m_i[e], mx);
            float alpha = __expf(m_i[e] - nm);
            float p0 = __expf(ps[0][e] - nm);
            float p1 = __expf(ps[1][e] - nm);
            float rs = p0 + p1;
            for (int off = 1; off < 16; off <<= 1)
                rs += __shfl_xor(rs, off);
            l_i[e] = l_i[e] * alpha + rs;
            m_i[e] = nm;
#pragma unroll
            for (int nt = 0; nt < 4; ++nt) oacc[nt][e] *= alpha;
            int row = e + 8 * half;
            pbuf[row * LDSTRIDE + lr]      = (__bf16)p0;
            pbuf[row * LDSTRIDE + 16 + lr] = (__bf16)p1;
        }
        // same-wave LDS RAW: wait on DS counter, then reload P in A-layout
        asm volatile("s_wait_dscnt 0" ::: "memory");
        v16bf ap = load_frag(pbuf + lr * LDSTRIDE + half * 8);

        // O += P(16x32) * V(32x64); V stored transposed so B rows are contiguous
#pragma unroll
        for (int nt = 0; nt < 4; ++nt)
            oacc[nt] = wmma_bf16(ap, vf[nt], oacc[nt]);

#pragma unroll
        for (int i = 0; i < 4; ++i) kf[i] = kn[i];
    }

    // normalize and write O into [B,T,C] (heads concatenated)
#pragma unroll
    for (int e = 0; e < 8; ++e) {
        float inv = 1.0f / l_i[e];
        int qi = wq0 + e + 8 * half;
        size_t orow = ((size_t)b * T_ + qi) * C_ + h * D_;
#pragma unroll
        for (int nt = 0; nt < 4; ++nt)
            ob[orow + nt * 16 + lr] = (__bf16)(oacc[nt][e] * inv);
    }
}

// ---------------- output projection ----------------
// grid: (BT/64, C/64)  block: 128.  Same staged/double-buffered structure.
__global__ __launch_bounds__(128) void oproj_kernel(
    const __bf16* __restrict__ ob, const __bf16* __restrict__ wpT,
    const float* __restrict__ bp, float* __restrict__ out)
{
    __shared__ __align__(16) __bf16 lds_b[2 * 64 * LDSTRIDE];

    const int tid  = threadIdx.x;
    const int wave = tid >> 5;
    const int lane = tid & 31;
    const int half = lane >> 4;
    const int lr   = lane & 15;
    const int rbase = blockIdx.x * 64 + wave * 16;
    const int n0    = blockIdx.y * 64;

    const __bf16* arow  = ob + (size_t)(rbase + lr) * C_ + half * 8;
    const __bf16* wbase = wpT + (size_t)n0 * C_;

    v8f acc[4] = {};

    stage_tile(lds_b, wbase, tid);
    v16bf a_cur = load_frag(arow);

    for (int k0 = 0; k0 < C_; k0 += 32) {
        const int cb = (k0 >> 5) & 1;
        wait_stage();
        __syncthreads();
        if (k0 + 32 < C_)
            stage_tile(lds_b + (cb ^ 1) * (64 * LDSTRIDE), wbase + k0 + 32, tid);
        v16bf a_nxt = a_cur;
        if (k0 + 32 < C_) a_nxt = load_frag(arow + k0 + 32);

        const __bf16* lb = lds_b + cb * (64 * LDSTRIDE);
#pragma unroll
        for (int nt = 0; nt < 4; ++nt) {
            v16bf b = load_frag(lb + (nt * 16 + lr) * LDSTRIDE + half * 8);
            acc[nt] = wmma_bf16(a_cur, b, acc[nt]);
        }
        a_cur = a_nxt;
    }

#pragma unroll
    for (int nt = 0; nt < 4; ++nt) {
        int col = n0 + nt * 16 + lr;
        float bias = bp[col];
#pragma unroll
        for (int e = 0; e < 8; ++e) {
            int row = rbase + e + 8 * half;
            out[(size_t)row * C_ + col] = acc[nt][e] + bias;
        }
    }
}

// ---------------- launcher ----------------

extern "C" void kernel_launch(void* const* d_in, const int* in_sizes, int n_in,
                              void* d_out, int out_size, void* d_ws, size_t ws_size,
                              hipStream_t stream) {
    (void)in_sizes; (void)n_in; (void)out_size; (void)ws_size;
    const float* x  = (const float*)d_in[0];
    const float* Wq = (const float*)d_in[1];
    const float* Wk = (const float*)d_in[2];
    const float* Wv = (const float*)d_in[3];
    const float* Wp = (const float*)d_in[4];
    const float* bp = (const float*)d_in[5];
    float* out = (float*)d_out;

    char* ws = (char*)d_ws;
    size_t off = 0;
    auto wsalloc = [&](size_t bytes) -> char* {
        char* p = ws + off;
        off += (bytes + 255) & ~(size_t)255;
        return p;
    };
    __bf16* xb  = (__bf16*)wsalloc((size_t)BT_ * C_ * 2);           // 16 MB
    __bf16* wqT = (__bf16*)wsalloc((size_t)H_ * D_ * C_ * 2);       //  2 MB
    __bf16* wkT = (__bf16*)wsalloc((size_t)H_ * D_ * C_ * 2);       //  2 MB
    __bf16* wvT = (__bf16*)wsalloc((size_t)H_ * D_ * C_ * 2);       //  2 MB
    __bf16* wpT = (__bf16*)wsalloc((size_t)C_ * C_ * 2);            //  2 MB
    __bf16* qb  = (__bf16*)wsalloc((size_t)B_ * H_ * T_ * D_ * 2);  // 16 MB
    __bf16* kb  = (__bf16*)wsalloc((size_t)B_ * H_ * T_ * D_ * 2);  // 16 MB
    __bf16* vtb = (__bf16*)wsalloc((size_t)B_ * H_ * T_ * D_ * 2 + 256); // 16 MB (+pad)
    __bf16* ob  = (__bf16*)wsalloc((size_t)BT_ * C_ * 2);           // 16 MB

    cvt_x_kernel<<<(BT_ * C_ + 255) / 256, 256, 0, stream>>>(x, xb, BT_ * C_);
    cvt_whcd_kernel<<<(H_ * C_ * D_) / 256, 256, 0, stream>>>(Wq, wqT);
    cvt_whcd_kernel<<<(H_ * C_ * D_) / 256, 256, 0, stream>>>(Wk, wkT);
    cvt_whcd_kernel<<<(H_ * C_ * D_) / 256, 256, 0, stream>>>(Wv, wvT);
    cvt_wp_kernel<<<(C_ * C_) / 256, 256, 0, stream>>>(Wp, wpT);

    qkv_kernel<<<dim3(BT_ / 64, H_, 3), 128, 0, stream>>>(xb, wqT, wkT, wvT, qb, kb, vtb);
    flash_kernel<<<dim3(T_ / 64, H_, B_), 128, 0, stream>>>(qb, kb, vtb, ob);
    oproj_kernel<<<dim3(BT_ / 64, C_ / 64), 128, 0, stream>>>(ob, wpT, bp, out);
}
